// DecoderRNN_70720931496163
// MI455X (gfx1250) — matrix operand order repaired and checked by reading, not compiled, not run on hardware
//
#include <hip/hip_runtime.h>

// DecoderRNN step on MI455X (gfx1250): bf16 WMMA GEMMs + fused GRU gates.
// Roofline: 60 GFLOP vs 256 MB HBM => compute-bound in fp32; bf16 WMMA
// (v_wmma_f32_16x16x32_bf16, fp32 accumulate) is the fastest adequate path.

#define HD        256
#define TILE_B    64
#define NTHREADS  256   // 8 waves (wave32)

typedef __attribute__((ext_vector_type(8)))  __bf16 bf16x8;
typedef __attribute__((ext_vector_type(16))) __bf16 bf16x16;
typedef __attribute__((ext_vector_type(8)))  float  v8f;

__device__ __forceinline__ v8f wmma_bf16(bf16x16 a, bf16x16 b, v8f c) {
    // (neg_a, A, neg_b, B, c_mod, C, reuse_a, reuse_b)
    return __builtin_amdgcn_wmma_f32_16x16x32_bf16(false, a, false, b,
                                                   (short)0, c, false, false);
}

// A-matrix 16x32 bf16 lane layout: lane<16 holds K{0..7,16..23} of row (lane),
// lane>=16 holds K{8..15,24..31}.  Caller pre-offsets p by (lane>>4)*8.
__device__ __forceinline__ bf16x16 load_a16(const __bf16* p) {
    bf16x8 lo = *(const bf16x8*)(p);
    bf16x8 hi = *(const bf16x8*)(p + 16);
    return __builtin_shufflevector(lo, hi, 0,1,2,3,4,5,6,7,8,9,10,11,12,13,14,15);
}

// B-matrix 32x16 bf16: lane n (mod 16) holds K{0..7,16..23} (lane<16) or
// K{8..15,24..31} (lane>=16) of column n.  Caller pre-offsets p by (lane>>4)*8.
__device__ __forceinline__ bf16x16 load_b16(const __bf16* p) {
    bf16x8 lo = *(const bf16x8*)(p);
    bf16x8 hi = *(const bf16x8*)(p + 16);
    return __builtin_shufflevector(lo, hi, 0,1,2,3,4,5,6,7,8,9,10,11,12,13,14,15);
}

__device__ __forceinline__ float sigm(float x) { return 1.0f / (1.0f + __expf(-x)); }

struct HTiles { v8f t[8]; };

// One GRU layer for this wave's 16-row x 128-col slab.
// Reads xb (bf16 x) and hb (bf16 h_prev) from LDS; writes h_new to global hdst
// and returns the 8 h_new tiles in registers (x update deferred past barrier).
__device__ __forceinline__ HTiles gru_layer(
    const __bf16* xb, const __bf16* hb,
    const __bf16* __restrict__ wi, const __bf16* __restrict__ wh,  // [768,256] bf16
    const float*  __restrict__ bi, const float*  __restrict__ bh,  // [768]
    const float*  __restrict__ hsrc,                               // [B,256] h_in
    float*        __restrict__ hdst,                               // [B,256] h_out
    int row0, int rb, int j0, int lm, int lh)
{
    HTiles hn;
    const __bf16* ax = xb + (rb * 16 + lm) * HD + lh * 8;
    const __bf16* ah = hb + (rb * 16 + lm) * HD + lh * 8;
#pragma unroll
    for (int jt = 0; jt < 8; ++jt) {
        const int j = j0 + jt * 16;
        v8f ir = {0,0,0,0,0,0,0,0}, iz = ir, in_ = ir;
        v8f hr = ir, hz = ir, hnm = ir;
        const __bf16* wir = wi + (size_t)(0 * HD + j + lm) * HD + lh * 8;
        const __bf16* wiz = wi + (size_t)(1 * HD + j + lm) * HD + lh * 8;
        const __bf16* win = wi + (size_t)(2 * HD + j + lm) * HD + lh * 8;
        const __bf16* whr = wh + (size_t)(0 * HD + j + lm) * HD + lh * 8;
        const __bf16* whz = wh + (size_t)(1 * HD + j + lm) * HD + lh * 8;
        const __bf16* whn = wh + (size_t)(2 * HD + j + lm) * HD + lh * 8;
#pragma unroll 2
        for (int k0 = 0; k0 < HD; k0 += 32) {
            bf16x16 a = load_a16(ax + k0);
            bf16x16 h = load_a16(ah + k0);
            ir  = wmma_bf16(a, load_b16(wir + k0), ir);
            iz  = wmma_bf16(a, load_b16(wiz + k0), iz);
            in_ = wmma_bf16(a, load_b16(win + k0), in_);
            hr  = wmma_bf16(h, load_b16(whr + k0), hr);
            hz  = wmma_bf16(h, load_b16(whz + k0), hz);
            hnm = wmma_bf16(h, load_b16(whn + k0), hnm);
        }
        const int col = j + lm;
        const float bir = bi[col], biz = bi[HD + col], bin = bi[2 * HD + col];
        const float bhr = bh[col], bhz = bh[HD + col], bhn = bh[2 * HD + col];
        v8f hv;
#pragma unroll
        for (int v = 0; v < 8; ++v) {
            const int grow = row0 + rb * 16 + v + 8 * lh;   // C/D: row = v + 8*(lane>>4)
            const float hold = hsrc[(size_t)grow * HD + col];
            const float r  = sigm(ir[v] + bir + hr[v] + bhr);
            const float z  = sigm(iz[v] + biz + hz[v] + bhz);
            const float nn = tanhf(in_[v] + bin + r * (hnm[v] + bhn));
            const float hnew = (1.0f - z) * nn + z * hold;
            hv[v] = hnew;
            hdst[(size_t)grow * HD + col] = hnew;
        }
        hn.t[jt] = hv;
    }
    return hn;
}

__global__ void cvt_f32_bf16(const float* __restrict__ src, __bf16* __restrict__ dst, int n) {
    int i = blockIdx.x * blockDim.x + threadIdx.x;
    if (i < n) dst[i] = (__bf16)src[i];
}

__global__ __launch_bounds__(NTHREADS)
void decoder_step(const float* __restrict__ attn_out,
                  const float* __restrict__ attn_hid,
                  const float* __restrict__ dec_hid,      // [2,B,H]
                  const __bf16* __restrict__ wlin,        // [256,512] bf16
                  const __bf16* __restrict__ w1i, const __bf16* __restrict__ w1h,
                  const float*  __restrict__ b1i, const float*  __restrict__ b1h,
                  const __bf16* __restrict__ w2i, const __bf16* __restrict__ w2h,
                  const float*  __restrict__ b2i, const float*  __restrict__ b2h,
                  float* __restrict__ out, int B)
{
    extern __shared__ char smem[];
    float*  xf  = (float*)(smem);                         // x f32   [64][256]  64 KB
    __bf16* xb  = (__bf16*)(smem + 65536);                // x bf16  [64][256]  32 KB
    __bf16* hb  = (__bf16*)(smem + 65536 + 32768);        // h bf16  [64][256]  32 KB
    __bf16* cat = (__bf16*)(smem + 65536 + 65536);        // cat bf16[64][512]  64 KB

    const int tid  = threadIdx.x;
    const int wave = tid >> 5, lane = tid & 31;
    const int lm = lane & 15, lh = lane >> 4;
    const int rb = wave & 3;                 // row block within tile (16 rows each)
    const int j0 = (wave >> 2) * 128;        // column half owned by this wave
    const int row0 = blockIdx.x * TILE_B;

    // ---- Stage cat = [attn_rnn_hidden | attn_out] and h0 as bf16 in LDS ----
    for (int i = tid; i < TILE_B * 512 / 4; i += NTHREADS) {
        const int e = i * 4, r = e >> 9, c = e & 511;
        const float4 v = (c < 256)
            ? *(const float4*)(attn_hid + (size_t)(row0 + r) * HD + c)
            : *(const float4*)(attn_out + (size_t)(row0 + r) * HD + (c - 256));
        __bf16* d = cat + r * 512 + c;
        d[0] = (__bf16)v.x; d[1] = (__bf16)v.y; d[2] = (__bf16)v.z; d[3] = (__bf16)v.w;
    }
    for (int i = tid; i < TILE_B * HD / 4; i += NTHREADS) {
        const int e = i * 4, r = e >> 8, c = e & 255;
        const float4 v = *(const float4*)(dec_hid + (size_t)(row0 + r) * HD + c);
        __bf16* d = hb + r * HD + c;
        d[0] = (__bf16)v.x; d[1] = (__bf16)v.y; d[2] = (__bf16)v.z; d[3] = (__bf16)v.w;
    }
    __syncthreads();

    // ---- Phase 0: x = cat @ W_lin^T  (K = 512) ----
    {
        const __bf16* arow = cat + (rb * 16 + lm) * 512 + lh * 8;
#pragma unroll
        for (int jt = 0; jt < 8; ++jt) {
            const int j = j0 + jt * 16;
            v8f acc = {0,0,0,0,0,0,0,0};
            const __bf16* brow = wlin + (size_t)(j + lm) * 512 + lh * 8;
#pragma unroll 2
            for (int k0 = 0; k0 < 512; k0 += 32)
                acc = wmma_bf16(load_a16(arow + k0), load_b16(brow + k0), acc);
#pragma unroll
            for (int v = 0; v < 8; ++v) {
                const int idx = (rb * 16 + v + 8 * lh) * HD + j + lm;
                xf[idx] = acc[v];
                xb[idx] = (__bf16)acc[v];
            }
        }
    }
    __syncthreads();

    // ---- GRU layer 1 ----
    HTiles h1t = gru_layer(xb, hb, w1i, w1h, b1i, b1h,
                           dec_hid, out + (size_t)B * HD,
                           row0, rb, j0, lm, lh);
    __syncthreads();                         // all reads of xb/hb complete

    // x += h0 (own slab), stage h1_in into hb
#pragma unroll
    for (int jt = 0; jt < 8; ++jt) {
        const int j = j0 + jt * 16;
#pragma unroll
        for (int v = 0; v < 8; ++v) {
            const int idx = (rb * 16 + v + 8 * lh) * HD + j + lm;
            const float xv = xf[idx] + h1t.t[jt][v];
            xf[idx] = xv;
            xb[idx] = (__bf16)xv;
        }
    }
    for (int i = tid; i < TILE_B * HD / 4; i += NTHREADS) {
        const int e = i * 4, r = e >> 8, c = e & 255;
        const float4 v = *(const float4*)(dec_hid + (size_t)B * HD +
                                          (size_t)(row0 + r) * HD + c);
        __bf16* d = hb + r * HD + c;
        d[0] = (__bf16)v.x; d[1] = (__bf16)v.y; d[2] = (__bf16)v.z; d[3] = (__bf16)v.w;
    }
    __syncthreads();

    // ---- GRU layer 2 ----
    HTiles h2t = gru_layer(xb, hb, w2i, w2h, b2i, b2h,
                           dec_hid + (size_t)B * HD, out + 2 * (size_t)B * HD,
                           row0, rb, j0, lm, lh);

    // ---- Final residual: out_x = x + h1, straight from registers ----
#pragma unroll
    for (int jt = 0; jt < 8; ++jt) {
        const int j = j0 + jt * 16;
#pragma unroll
        for (int v = 0; v < 8; ++v) {
            const int rr  = rb * 16 + v + 8 * lh;
            const int col = j + lm;
            out[(size_t)(row0 + rr) * HD + col] = xf[rr * HD + col] + h2t.t[jt][v];
        }
    }
}

extern "C" void kernel_launch(void* const* d_in, const int* in_sizes, int n_in,
                              void* d_out, int out_size, void* d_ws, size_t ws_size,
                              hipStream_t stream) {
    const float* attn_out = (const float*)d_in[0];
    const float* attn_hid = (const float*)d_in[1];
    const float* dec_hid  = (const float*)d_in[2];
    const float* W_lin    = (const float*)d_in[3];
    const float* g1_Wih   = (const float*)d_in[4];
    const float* g1_Whh   = (const float*)d_in[5];
    const float* g1_bih   = (const float*)d_in[6];
    const float* g1_bhh   = (const float*)d_in[7];
    const float* g2_Wih   = (const float*)d_in[8];
    const float* g2_Whh   = (const float*)d_in[9];
    const float* g2_bih   = (const float*)d_in[10];
    const float* g2_bhh   = (const float*)d_in[11];

    const int B = in_sizes[0] / HD;

    // bf16 weight cache in workspace (1.75 MB total)
    __bf16* wlin_b = (__bf16*)d_ws;
    __bf16* w1i_b  = wlin_b + 256 * 512;
    __bf16* w1h_b  = w1i_b  + 768 * 256;
    __bf16* w2i_b  = w1h_b  + 768 * 256;
    __bf16* w2h_b  = w2i_b  + 768 * 256;

    const int nlin = 256 * 512, ngru = 768 * 256;
    cvt_f32_bf16<<<(nlin + 255) / 256, 256, 0, stream>>>(W_lin,  wlin_b, nlin);
    cvt_f32_bf16<<<(ngru + 255) / 256, 256, 0, stream>>>(g1_Wih, w1i_b,  ngru);
    cvt_f32_bf16<<<(ngru + 255) / 256, 256, 0, stream>>>(g1_Whh, w1h_b,  ngru);
    cvt_f32_bf16<<<(ngru + 255) / 256, 256, 0, stream>>>(g2_Wih, w2i_b,  ngru);
    cvt_f32_bf16<<<(ngru + 255) / 256, 256, 0, stream>>>(g2_Whh, w2h_b,  ngru);

    const size_t shmem = 65536 + 32768 + 32768 + 65536;   // 192 KB (<320 KB/WGP)
    decoder_step<<<dim3(B / TILE_B), dim3(NTHREADS), shmem, stream>>>(
        attn_out, attn_hid, dec_hid, wlin_b,
        w1i_b, w1h_b, g1_bih, g1_bhh,
        w2i_b, w2h_b, g2_bih, g2_bhh,
        (float*)d_out, B);
}
